// WeightDemodulation_68367289417881
// MI455X (gfx1250) — matrix operand order repaired and checked by reading, not compiled, not run on hardware
//
#include <hip/hip_runtime.h>

typedef __bf16 bf16_t;
typedef __attribute__((ext_vector_type(16))) __bf16 v16bf;
typedef __attribute__((ext_vector_type(8)))  float  v8f;

#define HW   256
#define CIN  128
#define COUT 128
#define NB   8
#define TILE 16
#define HALO 18
#define PIXD 36   // dwords per LDS pixel row (72 bf16 = 64 data + 8 pad; multiple of 4 for b128)

// ---------------------------------------------------------------------------
// Kernel 1: gamma = y@gamma_w^T + gamma_b ; modulate conv_w ; demodulate ;
// emit bf16 weights laid out [b][cout][tap][cin] for tap-major K in the GEMM.
// ---------------------------------------------------------------------------
__global__ __launch_bounds__(128)
void modw_kernel(const float* __restrict__ y, const float* __restrict__ conv_w,
                 const float* __restrict__ gamma_w, const float* __restrict__ gamma_b,
                 bf16_t* __restrict__ wm)
{
  const int co = blockIdx.x;   // 0..127
  const int b  = blockIdx.y;   // 0..7
  const int ci = threadIdx.x;  // 0..127
  __shared__ float red[128];

  float g = gamma_b[ci];
  const float* yb = y + b * CIN;
  const float* gw = gamma_w + (size_t)ci * CIN;
  #pragma unroll 4
  for (int j = 0; j < CIN; ++j) g = fmaf(yb[j], gw[j], g);
  g += 1.0f;

  float wv[9];
  float ss = 0.0f;
  const float* cw = conv_w + ((size_t)co * CIN + ci) * 9;
  #pragma unroll
  for (int t = 0; t < 9; ++t) { float v = cw[t] * g; wv[t] = v; ss = fmaf(v, v, ss); }

  red[ci] = ss;
  __syncthreads();
  #pragma unroll
  for (int s = 64; s > 0; s >>= 1) {
    if (ci < s) red[ci] += red[ci + s];
    __syncthreads();
  }
  const float d = rsqrtf(red[0] + 1e-8f);

  bf16_t* o = wm + (((size_t)b * COUT + co) * 9) * CIN + ci;
  #pragma unroll
  for (int t = 0; t < 9; ++t) o[t * CIN] = (bf16_t)(wv[t] * d);
}

// ---------------------------------------------------------------------------
// Kernel 2: implicit-GEMM 3x3 SAME conv (groups = batch) via bf16 WMMA.
// WG = 256 threads (8 wave32) computes out[b][0:128][ty*16:+16][tx*16:+16].
// M = cout(128), N = 256 pixels, K = cin*9 (tap-major).
// x streamed with NT loads, out with NT stores -> L2 kept for the 2.3MB
// modulated-weight tensor that all 2048 WGs re-read.
// ---------------------------------------------------------------------------
__global__ __launch_bounds__(256)
void conv_wmma_kernel(const float* __restrict__ x, const bf16_t* __restrict__ wm,
                      float* __restrict__ out)
{
  __shared__ unsigned int lds_x[HALO * HALO * PIXD]; // 46,656 B

  const int b    = blockIdx.z;
  const int tx   = blockIdx.x;
  const int ty   = blockIdx.y;
  const int tid  = threadIdx.x;
  const int lane = tid & 31;
  const int wave = tid >> 5;      // 0..7
  const int m0   = wave * 16;     // this wave's cout slab
  const int hi   = lane >> 4;     // lane half (ISA fragment layouts)
  const int nlo  = lane & 15;

  v8f c[16];
  #pragma unroll
  for (int nt = 0; nt < 16; ++nt) c[nt] = (v8f){0.f,0.f,0.f,0.f,0.f,0.f,0.f,0.f};

  const int x0 = tx * TILE - 1;
  const int y0 = ty * TILE - 1;
  const float*  xb   = x + (size_t)b * CIN * HW * HW;
  const bf16_t* wrow = wm + (((size_t)b * COUT + (m0 + nlo)) * 9) * CIN; // A: M = lane&15

  for (int half = 0; half < 2; ++half) {   // two 64-channel rounds through LDS
    if (half) __syncthreads();
    const int cbase = half * 64;

    // Cooperative halo-tile load: f32 global -> bf16 LDS, [pixel][cin] layout.
    // 64*18*18 = 20736 = 81 * 256 -> no tail divergence (EXEC all-ones for WMMA).
    for (int e = tid; e < 64 * HALO * HALO; e += 256) {
      const int px = e % HALO;
      const int t2 = e / HALO;
      const int py = t2 % HALO;
      const int ci = t2 / HALO;
      const int gx = x0 + px;
      const int gy = y0 + py;
      float v = 0.0f;
      if ((unsigned)gx < HW && (unsigned)gy < HW)
        v = __builtin_nontemporal_load(&xb[((size_t)(cbase + ci) * HW + gy) * HW + gx]);
      ((bf16_t*)lds_x)[(py * HALO + px) * (PIXD * 2) + ci] = (bf16_t)v;
    }
    __syncthreads();

    for (int tap = 0; tap < 9; ++tap) {
      const int dy = tap / 3;
      const int dx = tap - dy * 3;
      #pragma unroll
      for (int kc = 0; kc < 2; ++kc) {     // K chunks of 32 within this 64-ch round
        // ---- A fragment (16x32 bf16), ISA layout: lanes0-15 K={0..7,16..23},
        // lanes16-31 K={8..15,24..31}; two b128 loads straight from L2.
        const unsigned int* ap = (const unsigned int*)(wrow + tap * CIN) +
                                 (half * 32 + kc * 16 + hi * 4);
        union { uint4 q[2]; v16bf v; } ua;
        ua.q[0] = *(const uint4*)(ap);
        ua.q[1] = *(const uint4*)(ap + 8);

        // ---- 16 N-tiles: B fragment (32x16 bf16) from LDS, K-pairs per dword;
        // lanes0-15 K={0..7,16..23}, lanes16-31 K={8..15,24..31}; two ds_load_b128 each.
        #pragma unroll
        for (int nt = 0; nt < 16; ++nt) {
          const unsigned int* bp = lds_x +
              ((nt + dy) * HALO + (nlo + dx)) * PIXD + kc * 16 + hi * 4;
          union { uint4 q[2]; v16bf v; } ub;
          ub.q[0] = *(const uint4*)(bp);
          ub.q[1] = *(const uint4*)(bp + 8);
          c[nt] = __builtin_amdgcn_wmma_f32_16x16x32_bf16(
                      false, ua.v, false, ub.v, (short)0, c[nt], false, false);
        }
      }
    }
  }

  // ---- Store C (ISA layout: VGPR r -> M = r + 8*hi, N = lane&15) ----
  float* ob = out + (size_t)b * COUT * HW * HW;
  const int coB = m0 + hi * 8;
  const int xo  = tx * TILE + nlo;
  #pragma unroll
  for (int nt = 0; nt < 16; ++nt) {
    const int yo = ty * TILE + nt;
    #pragma unroll
    for (int r = 0; r < 8; ++r)
      __builtin_nontemporal_store(c[nt][r],
          &ob[((size_t)(coB + r) * HW + yo) * HW + xo]);
  }
}

// ---------------------------------------------------------------------------
extern "C" void kernel_launch(void* const* d_in, const int* in_sizes, int n_in,
                              void* d_out, int out_size, void* d_ws, size_t ws_size,
                              hipStream_t stream) {
  (void)in_sizes; (void)n_in; (void)out_size; (void)ws_size;
  const float* x       = (const float*)d_in[0];
  const float* y       = (const float*)d_in[1];
  const float* conv_w  = (const float*)d_in[2];
  const float* gamma_w = (const float*)d_in[3];
  const float* gamma_b = (const float*)d_in[4];
  float* out = (float*)d_out;
  bf16_t* wm = (bf16_t*)d_ws;   // 8*128*9*128 bf16 = 2,359,296 B scratch

  modw_kernel<<<dim3(COUT, NB, 1), 128, 0, stream>>>(y, conv_w, gamma_w, gamma_b, wm);
  conv_wmma_kernel<<<dim3(HW / TILE, HW / TILE, NB), 256, 0, stream>>>(x, wm, out);
}
